// Cheb_35888746725726
// MI455X (gfx1250) — matrix operand order repaired and checked by reading, not compiled, not run on hardware
//
#include <hip/hip_runtime.h>
#include <hip/hip_bf16.h>

// ---------------------------------------------------------------------------
// ChebConv (K=3) x2 + linear head on MI455X (gfx1250, wave32).
// Dense GEMMs -> v_wmma_f32_16x16x32_bf16, one wave per 16-row stripe
// computing all six 16x16 column tiles (18 WMMA issues / wave, 6 independent
// accumulation chains). Weights pre-packed into WMMA per-lane operand order
// so B loads are contiguous 32B per lane. SpMM -> fp32 atomics, L2-resident.
// ---------------------------------------------------------------------------

#define NNODES 50000
#define NEDGES 800000
#define FDIM   96
#define NF     (NNODES * FDIM)
#define TILEM  (NNODES / 16)   // 3125
#define TILEN  (FDIM / 16)     // 6
#define WSLICE (FDIM * FDIM)   // 9216 elements per W[k] slice (packed same size)

typedef __bf16 bf16_t;
typedef __attribute__((ext_vector_type(16))) __bf16 v16bf;
typedef __attribute__((ext_vector_type(8)))  __bf16 v8bf;
typedef __attribute__((ext_vector_type(8)))  float  v8f;

union APack { v8bf h[2]; v16bf v; };

// ---------------- elementwise / setup kernels ------------------------------

__global__ void zero_f32(float* __restrict__ p, int n) {
    int i = blockIdx.x * blockDim.x + threadIdx.x;
    if (i < n) p[i] = 0.0f;
}

__global__ void deg_accum(const int* __restrict__ src,
                          const float* __restrict__ ew,
                          float* __restrict__ deg, int e) {
    int i = blockIdx.x * blockDim.x + threadIdx.x;
    if (i < e) atomicAdd(&deg[src[i]], ew[i]);
}

__global__ void deg_rsqrt(const float* __restrict__ deg,
                          float* __restrict__ dis, int n) {
    int i = blockIdx.x * blockDim.x + threadIdx.x;
    if (i < n) {
        float d = deg[i];
        dis[i] = (d > 0.0f) ? rsqrtf(fmaxf(d, 1e-30f)) : 0.0f;
    }
}

__global__ void edge_norm(const int* __restrict__ src,
                          const int* __restrict__ dst,
                          const float* __restrict__ ew,
                          const float* __restrict__ dis,
                          float* __restrict__ nw, int e) {
    int i = blockIdx.x * blockDim.x + threadIdx.x;
    if (i < e) {
        // (2/lambda_max) == 1 for lambda_max = 2
        nw[i] = -dis[src[i]] * ew[i] * dis[dst[i]];
    }
}

__global__ void f32_to_bf16(const float* __restrict__ x,
                            bf16_t* __restrict__ y, int n) {
    int i = blockIdx.x * blockDim.x + threadIdx.x;
    if (i < n) y[i] = (bf16_t)x[i];
}

// Repack W (K x 96 x 96 f32, row-major) into WMMA B-operand order, bf16.
// Packed index per slice: ((kc*6 + tn)*32 + lane)*16 + t
//   maps element  B[k][n]  with  k = kc*32 + (t>>3)*16 + (lane>>4)*8 + (t&7),
//                                n = tn*16 + (lane&15).
__global__ void pack_W(const float* __restrict__ W, bf16_t* __restrict__ out,
                       int total /* = 3*WSLICE */) {
    int i = blockIdx.x * blockDim.x + threadIdx.x;
    if (i >= total) return;
    int slice = i / WSLICE;
    int rem   = i - slice * WSLICE;
    int kc    = rem / (TILEN * 32 * 16);          // 3072 per kc
    int rem2  = rem - kc * (TILEN * 32 * 16);
    int tn    = rem2 / (32 * 16);
    int rem3  = rem2 - tn * (32 * 16);
    int lane  = rem3 >> 4;
    int t     = rem3 & 15;
    int k     = kc * 32 + (t >> 3) * 16 + (lane >> 4) * 8 + (t & 7);
    int n     = tn * 16 + (lane & 15);
    out[i] = (bf16_t)W[(size_t)slice * WSLICE + (size_t)k * FDIM + n];
}

// Tx2 = 2*Tx2 - Tx0   (in place on Tx2)
__global__ void cheb_combine(float* __restrict__ t2,
                             const float* __restrict__ t0, int n) {
    int i = blockIdx.x * blockDim.x + threadIdx.x;
    if (i < n) t2[i] = 2.0f * t2[i] - t0[i];
}

__global__ void bias_relu(const float* __restrict__ acc,
                          const float* __restrict__ b,
                          float* __restrict__ h, int n) {
    int i = blockIdx.x * blockDim.x + threadIdx.x;
    if (i < n) {
        float v = acc[i] + b[i % FDIM];
        h[i] = v > 0.0f ? v : 0.0f;
    }
}

// ---------------- SpMM: y[dst] += nw * h[src] ------------------------------
// One wave32 per edge, 3 features per lane (96 = 32*3), coalesced gathers.
__global__ void spmm_edges(const int* __restrict__ src,
                           const int* __restrict__ dst,
                           const float* __restrict__ nw,
                           const float* __restrict__ h,
                           float* __restrict__ y, int e) {
    int gid  = blockIdx.x * blockDim.x + threadIdx.x;
    int edge = gid >> 5;
    int lane = threadIdx.x & 31;
    if (edge >= e) return;
    int   s = src[edge];
    int   d = dst[edge];
    float w = nw[edge];
    const float* hr = h + (size_t)s * FDIM;
    float*       yr = y + (size_t)d * FDIM;
#pragma unroll
    for (int t = 0; t < 3; ++t) {
        int f = lane + 32 * t;
        atomicAdd(&yr[f], w * hr[f]);
    }
}

// ---------------- WMMA GEMM: C (+)= A(Nx96,bf16) @ B(96x96,packed bf16) ----
// One wave per 16-row stripe; computes all 6 column tiles (full 16x96 row
// band). A loaded once per K-chunk (two aligned 16B loads/lane), B operands
// read as one contiguous 32B load/lane from the pre-packed buffer.
__global__ void wmma_gemm96(const bf16_t* __restrict__ A,
                            const bf16_t* __restrict__ Bp,
                            float* __restrict__ C,
                            int nTilesM, int beta) {
    int gid  = blockIdx.x * blockDim.x + threadIdx.x;
    int wave = gid >> 5;
    if (wave >= nTilesM) return;             // uniform per wave: EXEC all-ones
    int lane = threadIdx.x & 31;
    int half = lane >> 4;                    // 0: lanes 0-15, 1: lanes 16-31
    int r    = lane & 15;
    int row0 = wave * 16;

    const bf16_t* arow = A + (size_t)(row0 + r) * FDIM;
    const bf16_t* bl   = Bp + (size_t)lane * 16;

    v8f acc[TILEN] = {};
#pragma unroll
    for (int kc = 0; kc < 3; ++kc) {
        int kb = kc * 32;
        // A: 16-bit 16x32 layout -> elems 0..7 : K = kb + half*8 + (0..7)
        //                           elems 8..15: K = kb + 16 + half*8 + (0..7)
        APack au;
        au.h[0] = *(const v8bf*)(arow + kb + half * 8);
        au.h[1] = *(const v8bf*)(arow + kb + 16 + half * 8);
#pragma unroll
        for (int tn = 0; tn < TILEN; ++tn) {
            v16bf bv = *(const v16bf*)(bl + (size_t)((kc * TILEN + tn) * 32) * 16);
            acc[tn] = __builtin_amdgcn_wmma_f32_16x16x32_bf16(
                /*neg_a=*/false, au.v, /*neg_b=*/false, bv,
                /*c_mod=*/(short)0, acc[tn],
                /*reuse_a=*/false, /*reuse_b=*/false);
        }
    }
    // C/D f32 layout: VGPR v -> row = row0 + half*8 + v, col = tn*16 + r
    float* cbase = C + (size_t)(row0 + half * 8) * FDIM + r;
#pragma unroll
    for (int tn = 0; tn < TILEN; ++tn) {
        float* crow = cbase + tn * 16;
        if (beta) {
#pragma unroll
            for (int v = 0; v < 8; ++v) crow[(size_t)v * FDIM] += acc[tn][v];
        } else {
#pragma unroll
            for (int v = 0; v < 8; ++v) crow[(size_t)v * FDIM] = acc[tn][v];
        }
    }
}

// ---------------- final head: out[n,0..1] = h[n,:] @ Wlin + blin -----------
__global__ void head_linear(const float* __restrict__ h,
                            const float* __restrict__ Wlin,
                            const float* __restrict__ blin,
                            float* __restrict__ out, int n) {
    int i = blockIdx.x * blockDim.x + threadIdx.x;
    if (i >= n) return;
    const float* hr = h + (size_t)i * FDIM;
    float a0 = blin[0], a1 = blin[1];
#pragma unroll 8
    for (int f = 0; f < FDIM; ++f) {
        float hv = hr[f];
        a0 += hv * Wlin[f * 2 + 0];
        a1 += hv * Wlin[f * 2 + 1];
    }
    out[i * 2 + 0] = a0;
    out[i * 2 + 1] = a1;
}

// ---------------------------------------------------------------------------

extern "C" void kernel_launch(void* const* d_in, const int* in_sizes, int n_in,
                              void* d_out, int out_size, void* d_ws, size_t ws_size,
                              hipStream_t stream) {
    (void)in_sizes; (void)n_in; (void)out_size; (void)ws_size;

    const float* x    = (const float*)d_in[0];
    const int*   ei   = (const int*)  d_in[1];
    const float* ew   = (const float*)d_in[2];
    const float* W1   = (const float*)d_in[3];
    const float* b1   = (const float*)d_in[4];
    const float* W2   = (const float*)d_in[5];
    const float* b2   = (const float*)d_in[6];
    const float* Wlin = (const float*)d_in[7];
    const float* blin = (const float*)d_in[8];
    float* out = (float*)d_out;

    const int* src = ei;
    const int* dst = ei + NEDGES;

    // ---- workspace carve-up (256B aligned) ----
    size_t cur = 0;
    auto take = [&](size_t bytes) -> void* {
        void* p = (char*)d_ws + cur;
        cur += (bytes + 255) & ~(size_t)255;
        return p;
    };
    float*  deg    = (float*) take((size_t)NNODES * 4);
    float*  dis    = (float*) take((size_t)NNODES * 4);
    float*  nw     = (float*) take((size_t)NEDGES * 4);
    float*  Tx1    = (float*) take((size_t)NF * 4);
    float*  Tx2    = (float*) take((size_t)NF * 4);
    float*  outacc = (float*) take((size_t)NF * 4);
    float*  hbuf   = (float*) take((size_t)NF * 4);   // layer-1 activations
    float*  h2buf  = (float*) take((size_t)NF * 4);   // layer-2 activations
    bf16_t* abuf   = (bf16_t*)take((size_t)NF * 2);
    bf16_t* W1bf   = (bf16_t*)take((size_t)3 * WSLICE * 2);  // packed
    bf16_t* W2bf   = (bf16_t*)take((size_t)3 * WSLICE * 2);  // packed

    const int B256  = 256;
    const int gN    = (NNODES + B256 - 1) / B256;
    const int gE    = (NEDGES + B256 - 1) / B256;
    const int gNF   = (NF + B256 - 1) / B256;
    const int gW    = (3 * WSLICE + B256 - 1) / B256;
    const int gGemm = (TILEM * 32 + B256 - 1) / B256;          // 8 waves/block
    const int gSpmm = (int)(((size_t)NEDGES * 32 + B256 - 1) / B256);

    // ---- graph normalization ----
    zero_f32 <<<gN, B256, 0, stream>>>(deg, NNODES);
    deg_accum<<<gE, B256, 0, stream>>>(src, ew, deg, NEDGES);
    deg_rsqrt<<<gN, B256, 0, stream>>>(deg, dis, NNODES);
    edge_norm<<<gE, B256, 0, stream>>>(src, dst, ew, dis, nw, NEDGES);

    // ---- weights: convert + repack into WMMA operand order ----
    pack_W<<<gW, B256, 0, stream>>>(W1, W1bf, 3 * WSLICE);
    pack_W<<<gW, B256, 0, stream>>>(W2, W2bf, 3 * WSLICE);

    // ---- one ChebConv layer ----
    auto cheb_layer = [&](const float* in, const bf16_t* Wbf,
                          const float* bias, float* hout) {
        // k = 0 : out = T0 @ W[0]
        f32_to_bf16<<<gNF, B256, 0, stream>>>(in, abuf, NF);
        wmma_gemm96<<<gGemm, B256, 0, stream>>>(abuf, Wbf, outacc, TILEM, 0);
        // T1 = L_hat @ T0
        zero_f32   <<<gNF, B256, 0, stream>>>(Tx1, NF);
        spmm_edges <<<gSpmm, B256, 0, stream>>>(src, dst, nw, in, Tx1, NEDGES);
        // out += T1 @ W[1]
        f32_to_bf16<<<gNF, B256, 0, stream>>>(Tx1, abuf, NF);
        wmma_gemm96<<<gGemm, B256, 0, stream>>>(abuf, Wbf + WSLICE, outacc, TILEM, 1);
        // T2 = 2 * (L_hat @ T1) - T0
        zero_f32    <<<gNF, B256, 0, stream>>>(Tx2, NF);
        spmm_edges  <<<gSpmm, B256, 0, stream>>>(src, dst, nw, Tx1, Tx2, NEDGES);
        cheb_combine<<<gNF, B256, 0, stream>>>(Tx2, in, NF);
        // out += T2 @ W[2]
        f32_to_bf16<<<gNF, B256, 0, stream>>>(Tx2, abuf, NF);
        wmma_gemm96<<<gGemm, B256, 0, stream>>>(abuf, Wbf + 2 * WSLICE, outacc, TILEM, 1);
        // h = relu(out + b)
        bias_relu<<<gNF, B256, 0, stream>>>(outacc, bias, hout, NF);
    };

    cheb_layer(x,    W1bf, b1, hbuf);
    cheb_layer(hbuf, W2bf, b2, h2buf);

    head_linear<<<gN, B256, 0, stream>>>(h2buf, Wlin, blin, out, NNODES);
}
